// ClientModel_90778428768465
// MI455X (gfx1250) — matrix operand order repaired and checked by reading, not compiled, not run on hardware
//
#include <hip/hip_runtime.h>
#include <hip/hip_fp16.h>

// Problem dims (match reference)
#define VSZ 10000
#define ED  512
#define HD  512
#define BB  64
#define TT  256
#define BT  (BB * TT)        // 16384
#define G3H (3 * HD)         // 1536
#define KIN (ED + HD)        // 1024

static_assert(VSZ % 16 == 0, "V tiles");
static_assert(BT % 128 == 0, "M tiles");

typedef __attribute__((ext_vector_type(16))) _Float16 v16h;
typedef __attribute__((ext_vector_type(8)))  _Float16 v8h;
typedef __attribute__((ext_vector_type(8)))  float    v8f;

// ---------------------------------------------------------------------------
// WMMA helpers: D = A(16x32 f16) x B(32x16 f16) + C(16x16 f32)
// A layout (16-bit A 16x32, k order matches B): lane&15 = row M; lanes<16 hold
// K = 0..15 ; lanes>=16 hold K = 16..31.
// B layout (K x N, N-major "transposed" weights Wt[n][k]): lane&15 = col N;
// lanes<16 hold K=0..15, lanes>=16 hold K=16..31 (contiguous 32B per lane).
// C/D layout: col = lane&15 ; vgpr r -> row r (lanes<16) / row 8+r (lanes>=16).
// ---------------------------------------------------------------------------
__device__ __forceinline__ v8f wmma16(v16h a, v16h b, v8f c) {
  return __builtin_amdgcn_wmma_f32_16x16x32_f16(false, a, false, b, (short)0, c,
                                                false, false);
}

__device__ __forceinline__ v16h load_a_frag(const _Float16* A, int lda, int lane) {
  const int r    = lane & 15;
  const int koff = (lane >> 4) << 4;         // 0 or 16
  const _Float16* p = A + r * lda + koff;
  v8h lo = *(const v8h*)(p);                 // K = koff .. koff+7
  v8h hi = *(const v8h*)(p + 8);             // K = koff+8 .. koff+15
  v16h a;
#pragma unroll
  for (int i = 0; i < 8; ++i) { a[i] = lo[i]; a[8 + i] = hi[i]; }
  return a;
}

__device__ __forceinline__ v16h load_b_frag(const _Float16* Bt, int ldb, int lane) {
  const int c    = lane & 15;
  const int koff = (lane >> 4) << 4;         // 0 or 16
  return *(const v16h*)(Bt + c * ldb + koff); // 32 contiguous bytes
}

// CDNA5 async global->LDS copy, 16B per lane, tracked with ASYNCcnt.
__device__ __forceinline__ void async_lds_copy16(void* lds_dst, const void* gsrc) {
  asm volatile("global_load_async_to_lds_b128 %0, %1, off"
               :: "v"((unsigned int)(size_t)lds_dst), "v"(gsrc)
               : "memory");
}
__device__ __forceinline__ void async_wait0() {
  asm volatile("s_wait_asynccnt 0x0" ::: "memory");
}

// ---------------------------------------------------------------------------
// Prep kernels
// ---------------------------------------------------------------------------
__global__ void k_gather(const float* __restrict__ emb,
                         const int* __restrict__ feat,
                         _Float16* __restrict__ X) {
  int idx = blockIdx.x * blockDim.x + threadIdx.x;   // one thread per 4 elems
  const int total = BT * (ED / 4);
  if (idx >= total) return;
  int bt = idx / (ED / 4);
  int e  = (idx - bt * (ED / 4)) * 4;
  const float4 v = *(const float4*)(emb + (size_t)feat[bt] * ED + e);
  _Float16* p = X + (size_t)bt * ED + e;
  p[0] = (_Float16)v.x; p[1] = (_Float16)v.y;
  p[2] = (_Float16)v.z; p[3] = (_Float16)v.w;
}

// Wt[n][k] = (f16) W[k][n] ; W is [K][N] row-major. Coalesced writes.
__global__ void k_tcast(const float* __restrict__ W, _Float16* __restrict__ Wt,
                        int K, int N) {
  int idx = blockIdx.x * blockDim.x + threadIdx.x;
  if (idx >= K * N) return;
  int n = idx / K, k = idx - n * K;
  Wt[(size_t)n * K + k] = (_Float16)W[(size_t)k * N + n];
}

// ---------------------------------------------------------------------------
// Input-projection GEMM: G[bt][n] = A[bt][0..511] . Wt[n][0..511]
// one wave per 16x16 tile, K=512 (16 WMMA steps)
// ---------------------------------------------------------------------------
__global__ void k_proj(const _Float16* __restrict__ A, int lda,
                       const _Float16* __restrict__ Wt, int ldw,
                       _Float16* __restrict__ G) {
  const int lane = threadIdx.x & 31;
  const int wave = blockIdx.x * (blockDim.x >> 5) + (threadIdx.x >> 5);
  const int NT = G3H / 16;                    // 96
  const int mt = wave / NT, nt = wave - mt * NT;
  const _Float16* Ab = A + (size_t)(mt * 16) * lda;
  const _Float16* Bb = Wt + (size_t)(nt * 16) * ldw;
  v8f acc = {};
#pragma unroll 4
  for (int k0 = 0; k0 < ED; k0 += 32) {
    v16h a = load_a_frag(Ab + k0, lda, lane);
    v16h b = load_b_frag(Bb + k0, ldw, lane);
    acc = wmma16(a, b, acc);
  }
  const int col   = nt * 16 + (lane & 15);
  const int rbase = mt * 16 + ((lane >> 4) << 3);
#pragma unroll
  for (int r = 0; r < 8; ++r)
    G[(size_t)(rbase + r) * G3H + col] = (_Float16)acc[r];
}

// ---------------------------------------------------------------------------
// Grid-wide spin barrier (agent scope) for the persistent recurrence kernel
// ---------------------------------------------------------------------------
struct GBar { unsigned int cnt; unsigned int gen; };

__device__ __forceinline__ void grid_barrier(GBar* bar, unsigned int nwg) {
  __threadfence();          // make this block's global stores agent-visible
  __syncthreads();
  if (threadIdx.x == 0) {
    unsigned int g = __hip_atomic_load(&bar->gen, __ATOMIC_RELAXED,
                                       __HIP_MEMORY_SCOPE_AGENT);
    unsigned int a = __hip_atomic_fetch_add(&bar->cnt, 1u, __ATOMIC_ACQ_REL,
                                            __HIP_MEMORY_SCOPE_AGENT);
    if (a == nwg - 1u) {
      __hip_atomic_store(&bar->cnt, 0u, __ATOMIC_RELAXED,
                         __HIP_MEMORY_SCOPE_AGENT);
      __hip_atomic_fetch_add(&bar->gen, 1u, __ATOMIC_RELEASE,
                             __HIP_MEMORY_SCOPE_AGENT);
    } else {
      while (__hip_atomic_load(&bar->gen, __ATOMIC_ACQUIRE,
                               __HIP_MEMORY_SCOPE_AGENT) == g) {
        __builtin_amdgcn_s_sleep(2);
      }
    }
  }
  __syncthreads();
}

// ---------------------------------------------------------------------------
// CIFG-LSTM recurrence, persistent kernel: 16 blocks x 256 thr = 128 waves.
// Wave (m,c): batch tile m (16 rows), H column chunk c (16 cols). Its three
// gate tiles (i at n=c16, j at 512+c16, o at 1024+c16) share the accumulator
// layout, so the gate nonlinearity is lane-local. c/h state live in VGPRs.
// h exchanged via ping-pong global buffers + grid barrier each step.
// ---------------------------------------------------------------------------
__global__ void k_recur(const _Float16* __restrict__ G,
                        const _Float16* __restrict__ Wt,
                        const float* __restrict__ bias,
                        const int* __restrict__ lens,
                        _Float16* hbuf,                 // 2 * BB*HD halfs
                        _Float16* __restrict__ Hout,    // [BT][HD]
                        GBar* bar) {
  const int lane  = threadIdx.x & 31;
  const int wid   = blockIdx.x * (blockDim.x >> 5) + (threadIdx.x >> 5);
  const int m     = wid & 3;                  // batch tile 0..3
  const int c     = wid >> 2;                 // H chunk 0..31
  const int col   = c * 16 + (lane & 15);
  const int rhalf = (lane >> 4) << 3;
  const int row0  = m * 16 + rhalf;           // batch row base for 8 accs

  const float bi = bias[col], bj = bias[HD + col], bo = bias[2 * HD + col];
  int lenv[8];
#pragma unroll
  for (int r = 0; r < 8; ++r) lenv[r] = lens[row0 + r];

  // B-frag bases: gate tiles, h-part of weights (k = ED..ED+511)
  const _Float16* Bi = Wt + (size_t)(0 * HD + c * 16) * KIN + ED;
  const _Float16* Bj = Wt + (size_t)(1 * HD + c * 16) * KIN + ED;
  const _Float16* Bo = Wt + (size_t)(2 * HD + c * 16) * KIN + ED;

  float cs[8], hs[8];
#pragma unroll
  for (int r = 0; r < 8; ++r) { cs[r] = 0.f; hs[r] = 0.f; }

  // zero the step-0 read buffer slice owned by this wave
#pragma unroll
  for (int r = 0; r < 8; ++r)
    hbuf[(size_t)(row0 + r) * HD + col] = (_Float16)0.f;
  grid_barrier(bar, gridDim.x);

  for (int t = 0; t < TT; ++t) {
    const _Float16* hr = hbuf + (size_t)(t & 1) * (BB * HD);
    _Float16*       hw = hbuf + (size_t)((t + 1) & 1) * (BB * HD);

    v8f zi, zj, zo;
#pragma unroll
    for (int r = 0; r < 8; ++r) {
      const size_t gbt = (size_t)((row0 + r) * TT + t) * G3H;
      zi[r] = (float)G[gbt + col] + bi;
      zj[r] = (float)G[gbt + HD + col] + bj;
      zo[r] = (float)G[gbt + 2 * HD + col] + bo;
    }

    const _Float16* Ab = hr + (size_t)(m * 16) * HD;
#pragma unroll 2
    for (int k0 = 0; k0 < HD; k0 += 32) {
      v16h a  = load_a_frag(Ab + k0, HD, lane);
      v16h b0 = load_b_frag(Bi + k0, KIN, lane);
      zi = wmma16(a, b0, zi);
      v16h b1 = load_b_frag(Bj + k0, KIN, lane);
      zj = wmma16(a, b1, zj);
      v16h b2 = load_b_frag(Bo + k0, KIN, lane);
      zo = wmma16(a, b2, zo);
    }

#pragma unroll
    for (int r = 0; r < 8; ++r) {
      const int  b   = row0 + r;
      const bool act = (t < lenv[r]);
      const float f  = 1.f / (1.f + __expf(-zi[r]));       // coupled gate
      const float cn = f * cs[r] + (1.f - f) * tanhf(zj[r]);
      const float hn = tanhf(cn) / (1.f + __expf(-zo[r]));
      if (act) { cs[r] = cn; hs[r] = hn; }                  // copy-through
      hw[(size_t)b * HD + col]              = (_Float16)hs[r];
      Hout[(size_t)(b * TT + t) * HD + col] = (_Float16)(act ? hn : 0.f);
    }
    grid_barrier(bar, gridDim.x);
  }
}

// ---------------------------------------------------------------------------
// Logits GEMM fused with online log-softmax + NLL.
// 128 blocks x 8 waves; wave w owns M-tile (blockIdx*8+w) and sweeps ALL 625
// V-tiles. All 8 waves walk the same nt sequence -> B tiles shared via WGP$
// (8x less L2 traffic than one-M-tile-per-block). The block's 8 A-tiles
// (128 KB) are staged in LDS with CDNA5 async global->LDS copies (ASYNCcnt).
// Each wave owns complete online-softmax stats for its 16 rows; final merge
// is a fixed-order block sum (deterministic).
// ---------------------------------------------------------------------------
__global__ void k_logits(const _Float16* __restrict__ Hseq,
                         const _Float16* __restrict__ LwT,
                         const float* __restrict__ lb,
                         const int* __restrict__ labels,
                         float* __restrict__ blocksum) {
  __shared__ alignas(32) _Float16 As[8 * 16 * HD];   // 128 KB
  __shared__ float rowNll[8 * 16];

  const int tid  = threadIdx.x;
  const int lane = tid & 31;
  const int wv   = tid >> 5;                   // 0..7
  const int mt   = blockIdx.x * 8 + wv;        // this wave's row tile

  // async-stage 128 contiguous rows (8 tiles) of Hseq into LDS
  {
    const _Float16* src = Hseq + (size_t)(blockIdx.x * 128) * HD;
    for (int i = tid * 8; i < 8 * 16 * HD; i += blockDim.x * 8)
      async_lds_copy16(As + i, src + i);
    async_wait0();
  }
  __syncthreads();

  const int rhalf = (lane >> 4) << 3;
  int lbl[8];
#pragma unroll
  for (int r = 0; r < 8; ++r) lbl[r] = labels[mt * 16 + rhalf + r];

  float mx[8], sm[8], ll[8];
#pragma unroll
  for (int r = 0; r < 8; ++r) { mx[r] = -1e30f; sm[r] = 0.f; ll[r] = -1e30f; }

  const _Float16* Aw = As + (size_t)(wv * 16) * HD;  // this wave's A tile
  const int NT = VSZ / 16;                     // 625
  for (int nt = 0; nt < NT; ++nt) {
    const int n = nt * 16 + (lane & 15);
    const float bv = lb[n];
    v8f acc;
#pragma unroll
    for (int r = 0; r < 8; ++r) acc[r] = bv;   // seed with bias
    const _Float16* Bb = LwT + (size_t)(nt * 16) * HD;
    if (nt + 1 < NT) {                         // hint next B tile into caches
      const _Float16* nb = LwT + (size_t)((nt + 1) * 16) * HD;
      __builtin_prefetch(nb + (size_t)(lane & 15) * HD + ((lane >> 4) << 4), 0, 0);
    }
#pragma unroll 4
    for (int k0 = 0; k0 < HD; k0 += 32) {
      v16h a = load_a_frag(Aw + k0, HD, lane);
      v16h b = load_b_frag(Bb + k0, HD, lane);
      acc = wmma16(a, b, acc);
    }
#pragma unroll
    for (int r = 0; r < 8; ++r) {
      const float v  = acc[r];
      const float mn = fmaxf(mx[r], v);
      sm[r] = sm[r] * __expf(mx[r] - mn) + __expf(v - mn);
      mx[r] = mn;
      if (lbl[r] == n) ll[r] = v;
    }
  }

  // merge (max,sum,label) across the 16 lanes of each half
#pragma unroll
  for (int r = 0; r < 8; ++r) {
#pragma unroll
    for (int off = 1; off < 16; off <<= 1) {
      const float mo = __shfl_xor(mx[r], off, 32);
      const float so = __shfl_xor(sm[r], off, 32);
      const float lo = __shfl_xor(ll[r], off, 32);
      const float mn = fmaxf(mx[r], mo);
      sm[r] = sm[r] * __expf(mx[r] - mn) + so * __expf(mo - mn);
      mx[r] = mn;
      ll[r] = fmaxf(ll[r], lo);
    }
  }
  if ((lane & 15) == 0) {
#pragma unroll
    for (int r = 0; r < 8; ++r)                // -log softmax at label
      rowNll[wv * 16 + rhalf + r] = (mx[r] + logf(sm[r])) - ll[r];
  }
  __syncthreads();
  if (tid == 0) {
    float acc = 0.f;
    for (int i = 0; i < 8 * 16; ++i) acc += rowNll[i];   // fixed order
    blocksum[blockIdx.x] = acc;
  }
}

__global__ void k_final(const float* __restrict__ blocksum,
                        float* __restrict__ out) {
  if (threadIdx.x == 0 && blockIdx.x == 0) {
    float s = 0.f;
    for (int i = 0; i < BT / 128; ++i) s += blocksum[i]; // fixed order
    out[0] = s / (float)BT;
  }
}

__global__ void k_init(GBar* bar) {
  if (threadIdx.x == 0) { bar->cnt = 0u; bar->gen = 0u; }
}

// ---------------------------------------------------------------------------
extern "C" void kernel_launch(void* const* d_in, const int* in_sizes, int n_in,
                              void* d_out, int out_size, void* d_ws,
                              size_t ws_size, hipStream_t stream) {
  (void)in_sizes; (void)n_in; (void)out_size; (void)ws_size;
  const int*   feat   = (const int*)d_in[0];
  const int*   labels = (const int*)d_in[1];
  const int*   lens   = (const int*)d_in[2];
  const float* emb    = (const float*)d_in[3];
  const float* W0     = (const float*)d_in[4];
  const float* b0     = (const float*)d_in[5];
  const float* W1     = (const float*)d_in[6];
  const float* b1     = (const float*)d_in[7];
  const float* Lw     = (const float*)d_in[8];
  const float* Lb     = (const float*)d_in[9];
  float* out = (float*)d_out;

  char* ws = (char*)d_ws;
  size_t off = 0;
  auto carve = [&](size_t bytes) {
    char* p = ws + off;
    off = (off + bytes + 255) & ~(size_t)255;
    return p;
  };
  _Float16* X    = (_Float16*)carve((size_t)BT * ED * 2);     // 16 MB
  _Float16* W0t  = (_Float16*)carve((size_t)G3H * KIN * 2);   // 3 MB
  _Float16* W1t  = (_Float16*)carve((size_t)G3H * KIN * 2);   // 3 MB
  _Float16* LwT  = (_Float16*)carve((size_t)VSZ * HD * 2);    // 10 MB
  _Float16* G    = (_Float16*)carve((size_t)BT * G3H * 2);    // 50 MB
  _Float16* H1   = (_Float16*)carve((size_t)BT * HD * 2);     // 16 MB
  _Float16* H2   = (_Float16*)carve((size_t)BT * HD * 2);     // 16 MB
  _Float16* hbuf = (_Float16*)carve((size_t)2 * BB * HD * 2); // 128 KB
  GBar*     bar  = (GBar*)carve(256);
  float*    bsum = (float*)carve((size_t)(BT / 128) * 4);

  k_init<<<1, 32, 0, stream>>>(bar);
  { int n = BT * (ED / 4);
    k_gather<<<(n + 255) / 256, 256, 0, stream>>>(emb, feat, X); }
  { int n = KIN * G3H;
    k_tcast<<<(n + 255) / 256, 256, 0, stream>>>(W0, W0t, KIN, G3H); }
  { int n = KIN * G3H;
    k_tcast<<<(n + 255) / 256, 256, 0, stream>>>(W1, W1t, KIN, G3H); }
  { int n = HD * VSZ;
    k_tcast<<<(n + 255) / 256, 256, 0, stream>>>(Lw, LwT, HD, VSZ); }

  const int projBlocks = (BT / 16) * (G3H / 16) / 8;  // 1 wave per tile
  k_proj<<<projBlocks, 256, 0, stream>>>(X, ED, W0t, KIN, G);
  k_recur<<<16, 256, 0, stream>>>(G, W0t, b0, lens, hbuf, H1, bar);
  k_proj<<<projBlocks, 256, 0, stream>>>(H1, HD, W1t, KIN, G);
  k_recur<<<16, 256, 0, stream>>>(G, W1t, b1, lens, hbuf, H2, bar);
  k_logits<<<BT / 128, 256, 0, stream>>>(H2, LwT, Lb, labels, bsum);
  k_final<<<1, 32, 0, stream>>>(bsum, out);
}
